// DentateCircuit_6433861009980
// MI455X (gfx1250) — matrix operand order repaired and checked by reading, not compiled, not run on hardware
//
#include <hip/hip_runtime.h>
#include <stdint.h>

// ---------------- model constants ----------------
#define NGC   20000
#define NMC   600
#define NPV   600
#define NSST  400
#define NMEC  1200
#define BATCH 4
#define TSTEPS 100

constexpr float F_VREST = -70.0f;
constexpr float F_EINH  = -70.0f;
constexpr float F_THETA = -50.0f;
constexpr float F_SIGMA = 5.0f;
constexpr float F_DTM   = 0.005f;                    // DT / TAU_M
constexpr float F_DAMPA = 0.95122942450071400910f;   // exp(-0.1/2)
constexpr float F_DGABA = 0.99004983374916805357f;   // exp(-0.1/10)

// K chunks of 32 (pre dim), N tiles of 16 (post dim)
constexpr int KC_GC = 625, KC_MC = 19, KC_PV = 19, KC_SST = 13, KC_MEC = 38;
constexpr int NT_GC = 1250, NT_MC = 38, NT_PV = 38, NT_SST = 25;

// A-stream chunk offsets (rates packed in WMMA-A register image), 512 bf16 per chunk
constexpr int ACH_GC = 0, ACH_MC = 625, ACH_PV = 644, ACH_SST = 663, ACH_MEC = 676;
constexpr int ACH_TOT = 714;

// state float offsets within each of v / se / si regions
constexpr int SO_GC = 0, SO_MC = 4 * NGC, SO_PV = SO_MC + 4 * NMC,
              SO_SST = SO_PV + 4 * NPV, SO_TOT = SO_SST + 4 * NSST; // 86400

// packed bf16 weight offsets, in units of tile-chunks (512 bf16 = 1024 B each)
constexpr size_t WOFF_GC_MC  = 0;
constexpr size_t WOFF_GC_PV  = 23750;
constexpr size_t WOFF_GC_SST = 47500;
constexpr size_t WOFF_MC_GC  = 63125;
constexpr size_t WOFF_MC_PV  = 86875;
constexpr size_t WOFF_MC_SST = 87597;
constexpr size_t WOFF_MC_MC  = 88072;
constexpr size_t WOFF_MEC_GC = 88794;
constexpr size_t WOFF_MEC_PV = 136294;
constexpr size_t WOFF_PV_GC  = 137738;
constexpr size_t WOFF_PV_MC  = 161488;
constexpr size_t WOFF_PV_PV  = 162210;
constexpr size_t WOFF_SST_GC = 162932;
constexpr size_t WOFF_SST_MC = 179182;
constexpr size_t WOFF_SST_PV = 179676;
constexpr size_t WOFF_SST_SST= 180170;
constexpr size_t W_TC_TOTAL  = 180495;   // ~176.3 MiB bf16 -> L2 resident

// tile groups (4 post-tiles per wave) and K-slices for the heavy GC-pre GEMMs
constexpr int G_GC = 313;                 // ceil(1250/4)
constexpr int G_MC = 10, G_PV = 10, G_SST = 7;
constexpr int G_HEAVY = G_MC + G_PV + G_SST;   // 27
constexpr int NSLICE = 8;                 // K-slices over the 625 GC chunks
constexpr int SLICE_KC = 79;              // 7*79 + 72 = 625
constexpr int U1_BLOCKS = G_GC + G_HEAVY * NSLICE;   // 529
constexpr int U2_BLOCKS = NT_MC + NT_PV + NT_SST;    // 101
constexpr size_t PBUF_FLOATS = (size_t)G_HEAVY * NSLICE * 4 * 128; // 110592

typedef __attribute__((ext_vector_type(16))) __bf16 v16bf;
typedef __attribute__((ext_vector_type(8)))  float  v8f;

__device__ __forceinline__ int imin(int a, int b) { return a < b ? a : b; }

// ---------------- helpers ----------------
__device__ __forceinline__ uint16_t f2bf(float x) {
  union { float f; uint32_t u; } cv; cv.f = x;
  uint32_t u = cv.u;
  return (uint16_t)((u + 0x7FFFu + ((u >> 16) & 1u)) >> 16);  // RNE
}

__device__ __forceinline__ void store32B(uint16_t* dst, const uint16_t* v) {
  uint32_t w[8];
#pragma unroll
  for (int j = 0; j < 8; ++j)
    w[j] = (uint32_t)v[2 * j] | ((uint32_t)v[2 * j + 1] << 16);
  uint4* q = (uint4*)dst;
  q[0] = make_uint4(w[0], w[1], w[2], w[3]);
  q[1] = make_uint4(w[4], w[5], w[6], w[7]);
}

// 4-tile register-blocked bf16 WMMA accumulation over chunk range [c0,c1):
// one shared A fragment per chunk, 4 B streams, 4 accumulators.
__device__ __forceinline__ void wacc4(v8f acc[4],
                                      const uint16_t* __restrict__ W, size_t woff,
                                      int kcFull, int c0, int c1, const int lt[4],
                                      const uint16_t* __restrict__ As, int achunk0,
                                      int lane) {
  const uint16_t* a  = As + ((size_t)achunk0 + c0) * 512 + (size_t)lane * 16;
  const uint16_t* w0 = W + (woff + (size_t)lt[0] * kcFull + c0) * 512 + (size_t)lane * 16;
  const uint16_t* w1 = W + (woff + (size_t)lt[1] * kcFull + c0) * 512 + (size_t)lane * 16;
  const uint16_t* w2 = W + (woff + (size_t)lt[2] * kcFull + c0) * 512 + (size_t)lane * 16;
  const uint16_t* w3 = W + (woff + (size_t)lt[3] * kcFull + c0) * 512 + (size_t)lane * 16;
  for (int c = c0; c < c1; ++c) {
    v16bf av = *(const v16bf*)a;
    v16bf b0 = *(const v16bf*)w0;
    v16bf b1 = *(const v16bf*)w1;
    v16bf b2 = *(const v16bf*)w2;
    v16bf b3 = *(const v16bf*)w3;
    acc[0] = __builtin_amdgcn_wmma_f32_16x16x32_bf16(false, av, false, b0, (short)0, acc[0], false, false);
    acc[1] = __builtin_amdgcn_wmma_f32_16x16x32_bf16(false, av, false, b1, (short)0, acc[1], false, false);
    acc[2] = __builtin_amdgcn_wmma_f32_16x16x32_bf16(false, av, false, b2, (short)0, acc[2], false, false);
    acc[3] = __builtin_amdgcn_wmma_f32_16x16x32_bf16(false, av, false, b3, (short)0, acc[3], false, false);
    a += 512; w0 += 512; w1 += 512; w2 += 512; w3 += 512;
  }
}

// Trace decay + conductance-based membrane update for one (tile-lane) neuron.
__device__ __forceinline__ void state_update(float* __restrict__ state, int soff,
                                             int Npop, int n,
                                             const v8f& ae, const v8f& ai) {
  float* vst = state;
  float* se  = state + SO_TOT;
  float* si  = state + 2 * SO_TOT;
#pragma unroll
  for (int b = 0; b < BATCH; ++b) {
    size_t idx = (size_t)soff + (size_t)b * Npop + n;
    float s_e = se[idx] * F_DAMPA + ae[b];
    float s_i = si[idx] * F_DGABA + ai[b];
    float vv = vst[idx];
    vv += F_DTM * ((F_VREST - vv) - s_e * vv + s_i * (F_EINH - vv));  // E_EXC = 0
    se[idx] = s_e;
    si[idx] = s_i;
    vst[idx] = vv;
  }
}

// ---------------- kernels ----------------

// One-time: fp32 [K,N] -> bf16 tiles in the WMMA B-operand register image.
__global__ void dg_pack_w_kernel(const float* __restrict__ g,
                                 uint16_t* __restrict__ wp,
                                 int Kpre, int Npost, int kc) {
  int id = blockIdx.x;                 // tile * kc + chunk
  int tile = id / kc, c = id % kc;
  int lane = threadIdx.x;
  int n = tile * 16 + (lane & 15);
  int kbase = c * 32 + ((lane >= 16) ? 8 : 0);
  uint16_t vals[16];
#pragma unroll
  for (int e = 0; e < 16; ++e) {
    int k = kbase + ((e < 8) ? e : e + 8);
    float x = (k < Kpre && n < Npost) ? g[(size_t)k * Npost + n] : 0.0f;
    vals[e] = f2bf(x);
  }
  store32B(wp + (size_t)id * 512 + (size_t)lane * 16, vals);
}

__global__ void dg_init_state_kernel(float* __restrict__ state) {
  int i = blockIdx.x * blockDim.x + threadIdx.x;
  if (i < 3 * SO_TOT) state[i] = (i < SO_TOT) ? F_VREST : 0.0f;  // v | se | si
}

// Per step: rates for all populations, packed into the WMMA A-operand register
// image (lane%16 = M = batch, ISA 16-bit A layout K mapping). Also emits r_gc
// to d_out and packs the MEC input slice for step t (used raw, no sigmoid).
__global__ void dg_rates_pack_kernel(const float* __restrict__ state,
                                     const float* __restrict__ mec,
                                     uint16_t* __restrict__ As,
                                     float* __restrict__ out, int t) {
  int cid = blockIdx.x;
  int lane = threadIdx.x;
  int c, Kpop, soff;
  bool isMec = false, isGC = false;
  if (cid < ACH_MC)        { c = cid;            Kpop = NGC;  soff = SO_GC; isGC = true; }
  else if (cid < ACH_PV)   { c = cid - ACH_MC;   Kpop = NMC;  soff = SO_MC; }
  else if (cid < ACH_SST)  { c = cid - ACH_PV;   Kpop = NPV;  soff = SO_PV; }
  else if (cid < ACH_MEC)  { c = cid - ACH_SST;  Kpop = NSST; soff = SO_SST; }
  else                     { c = cid - ACH_MEC;  Kpop = NMEC; soff = 0; isMec = true; }

  int b = lane & 15;
  int khi = (lane >= 16) ? 8 : 0;
  uint16_t vals[16];
#pragma unroll
  for (int e = 0; e < 16; ++e) {
    // 16-bit A layout: lane<16 holds K={0..7,16..23}, lane>=16 holds K={8..15,24..31}
    int k = c * 32 + ((e < 8) ? e : e + 8) + khi;
    float r = 0.0f;
    if (b < BATCH && k < Kpop) {
      if (isMec) {
        r = mec[((size_t)b * TSTEPS + t) * NMEC + k];
      } else {
        float vv = state[soff + b * Kpop + k];
        r = 1.0f / (1.0f + __expf(-(vv - F_THETA) / F_SIGMA));
        if (isGC) out[((size_t)b * TSTEPS + t) * NGC + k] = r;
      }
    }
    vals[e] = f2bf(r);
  }
  store32B(As + (size_t)cid * 512 + (size_t)lane * 16, vals);
}

// Per step, pass 1: one wave per 4-tile group.
//  - GC groups (short K): fused e/i accumulation + direct state update.
//  - MC/PV/SST groups: split the 625-chunk GC-presynaptic K range into 8
//    slices; each slice-wave writes partial e/i accumulators to pbuf
//    (deterministic slot per (group, slice, tile) -- no float atomics).
__global__ void dg_step_u1_kernel(const uint16_t* __restrict__ W,
                                  const uint16_t* __restrict__ As,
                                  float* __restrict__ state,
                                  float* __restrict__ pbuf) {
  int bid = blockIdx.x;
  int lane = threadIdx.x;
  const v8f z = {0.f, 0.f, 0.f, 0.f, 0.f, 0.f, 0.f, 0.f};
  v8f acc_e[4] = {z, z, z, z};
  v8f acc_i[4] = {z, z, z, z};

  if (bid < G_GC) {
    int t0 = bid * 4;
    int lt[4];
#pragma unroll
    for (int j = 0; j < 4; ++j) lt[j] = imin(t0 + j, NT_GC - 1);
    wacc4(acc_e, W, WOFF_MC_GC,  KC_MC,  0, KC_MC,  lt, As, ACH_MC,  lane);
    wacc4(acc_e, W, WOFF_MEC_GC, KC_MEC, 0, KC_MEC, lt, As, ACH_MEC, lane);
    wacc4(acc_i, W, WOFF_PV_GC,  KC_PV,  0, KC_PV,  lt, As, ACH_PV,  lane);
    wacc4(acc_i, W, WOFF_SST_GC, KC_SST, 0, KC_SST, lt, As, ACH_SST, lane);
    if (lane < 16) {
#pragma unroll
      for (int j = 0; j < 4; ++j) {
        int tile = t0 + j;
        if (tile < NT_GC)
          state_update(state, SO_GC, NGC, tile * 16 + lane, acc_e[j], acc_i[j]);
      }
    }
    return;
  }

  int idx = bid - G_GC;
  int gAll = idx >> 3;
  int s = idx & 7;
  int g, nt, pop;
  size_t wgc;
  if (gAll < G_MC)            { g = gAll;                 nt = NT_MC;  wgc = WOFF_GC_MC;  pop = 1; }
  else if (gAll < G_MC + G_PV){ g = gAll - G_MC;          nt = NT_PV;  wgc = WOFF_GC_PV;  pop = 2; }
  else                        { g = gAll - (G_MC + G_PV); nt = NT_SST; wgc = WOFF_GC_SST; pop = 3; }
  int lt[4];
#pragma unroll
  for (int j = 0; j < 4; ++j) lt[j] = imin(g * 4 + j, nt - 1);

  int c0 = s * SLICE_KC;
  int c1 = imin(KC_GC, c0 + SLICE_KC);
  wacc4(acc_e, W, wgc, KC_GC, c0, c1, lt, As, ACH_GC, lane);

  if (pop == 2 && s == 6)  // put the MEC->PV segment on the lighter slice 6
    wacc4(acc_e, W, WOFF_MEC_PV, KC_MEC, 0, KC_MEC, lt, As, ACH_MEC, lane);

  if (s == 7) {            // short e-segments + all inhibitory segments
    if (pop == 1) {
      wacc4(acc_e, W, WOFF_MC_MC,  KC_MC,  0, KC_MC,  lt, As, ACH_MC,  lane);
      wacc4(acc_i, W, WOFF_PV_MC,  KC_PV,  0, KC_PV,  lt, As, ACH_PV,  lane);
      wacc4(acc_i, W, WOFF_SST_MC, KC_SST, 0, KC_SST, lt, As, ACH_SST, lane);
    } else if (pop == 2) {
      wacc4(acc_e, W, WOFF_MC_PV,  KC_MC,  0, KC_MC,  lt, As, ACH_MC,  lane);
      wacc4(acc_i, W, WOFF_PV_PV,  KC_PV,  0, KC_PV,  lt, As, ACH_PV,  lane);
      wacc4(acc_i, W, WOFF_SST_PV, KC_SST, 0, KC_SST, lt, As, ACH_SST, lane);
    } else {
      wacc4(acc_e, W, WOFF_MC_SST,  KC_MC,  0, KC_MC,  lt, As, ACH_MC,  lane);
      wacc4(acc_i, W, WOFF_SST_SST, KC_SST, 0, KC_SST, lt, As, ACH_SST, lane);
    }
  }

  if (lane < 16) {
    float* base = pbuf + ((size_t)(gAll * NSLICE + s) * 4) * 128;
#pragma unroll
    for (int j = 0; j < 4; ++j) {
#pragma unroll
      for (int b = 0; b < BATCH; ++b) {
        base[(size_t)j * 128 + b * 16 + lane]      = acc_e[j][b];
        base[(size_t)j * 128 + b * 16 + lane + 64] = acc_i[j][b];
      }
    }
  }
}

// Per step, pass 2: sum the 8 K-slice partials (fixed order -> deterministic)
// and apply the trace/membrane update for MC/PV/SST.
__global__ void dg_step_u2_kernel(const float* __restrict__ pbuf,
                                  float* __restrict__ state) {
  int bid = blockIdx.x;
  int lane = threadIdx.x;
  int tile, Npop, soff, gbase;
  if (bid < NT_MC)              { tile = bid;                 Npop = NMC;  soff = SO_MC;  gbase = 0; }
  else if (bid < NT_MC + NT_PV) { tile = bid - NT_MC;         Npop = NPV;  soff = SO_PV;  gbase = G_MC; }
  else                          { tile = bid - NT_MC - NT_PV; Npop = NSST; soff = SO_SST; gbase = G_MC + G_PV; }
  if (lane >= 16) return;
  int n = tile * 16 + lane;
  if (n >= Npop) return;
  int gAll = gbase + (tile >> 2);
  int j = tile & 3;

  float* vst = state;
  float* se  = state + SO_TOT;
  float* si  = state + 2 * SO_TOT;
#pragma unroll
  for (int b = 0; b < BATCH; ++b) {
    float e = 0.0f, iv = 0.0f;
#pragma unroll
    for (int s = 0; s < NSLICE; ++s) {
      const float* p = pbuf + (((size_t)gAll * NSLICE + s) * 4 + j) * 128 + b * 16 + lane;
      e  += p[0];
      iv += p[64];
    }
    size_t idx = (size_t)soff + (size_t)b * Npop + n;
    float s_e = se[idx] * F_DAMPA + e;
    float s_i = si[idx] * F_DGABA + iv;
    float vv = vst[idx];
    vv += F_DTM * ((F_VREST - vv) - s_e * vv + s_i * (F_EINH - vv));
    se[idx] = s_e;
    si[idx] = s_i;
    vst[idx] = vv;
  }
}

// ---------------- launcher ----------------
extern "C" void kernel_launch(void* const* d_in, const int* in_sizes, int n_in,
                              void* d_out, int out_size, void* d_ws, size_t ws_size,
                              hipStream_t stream) {
  (void)in_sizes; (void)n_in; (void)out_size; (void)ws_size;

  uint16_t* Wb = (uint16_t*)d_ws;                       // packed bf16 weights
  uint16_t* Ab = Wb + W_TC_TOTAL * 512;                 // packed rate A-streams
  float* state = (float*)(Ab + (size_t)ACH_TOT * 512);  // v | se | si
  float* pbuf  = state + 3 * SO_TOT;                    // K-slice partials

  struct MD { int idx; size_t off; int K, N, kc, nt; };
  const MD mats[16] = {
    { 0, WOFF_GC_MC,   NGC,  NMC,  KC_GC,  NT_MC  },
    { 1, WOFF_GC_PV,   NGC,  NPV,  KC_GC,  NT_PV  },
    { 2, WOFF_GC_SST,  NGC,  NSST, KC_GC,  NT_SST },
    { 3, WOFF_MC_GC,   NMC,  NGC,  KC_MC,  NT_GC  },
    { 4, WOFF_MC_PV,   NMC,  NPV,  KC_MC,  NT_PV  },
    { 5, WOFF_MC_SST,  NMC,  NSST, KC_MC,  NT_SST },
    { 6, WOFF_MC_MC,   NMC,  NMC,  KC_MC,  NT_MC  },
    { 7, WOFF_MEC_GC,  NMEC, NGC,  KC_MEC, NT_GC  },
    { 8, WOFF_MEC_PV,  NMEC, NPV,  KC_MEC, NT_PV  },
    { 9, WOFF_PV_GC,   NPV,  NGC,  KC_PV,  NT_GC  },
    {10, WOFF_PV_MC,   NPV,  NMC,  KC_PV,  NT_MC  },
    {11, WOFF_PV_PV,   NPV,  NPV,  KC_PV,  NT_PV  },
    {12, WOFF_SST_GC,  NSST, NGC,  KC_SST, NT_GC  },
    {13, WOFF_SST_MC,  NSST, NMC,  KC_SST, NT_MC  },
    {14, WOFF_SST_PV,  NSST, NPV,  KC_SST, NT_PV  },
    {15, WOFF_SST_SST, NSST, NSST, KC_SST, NT_SST },
  };

  // One-time bf16 pack + swizzle of all 16 conductance matrices (~0.55 GB traffic)
  for (int i = 0; i < 16; ++i) {
    int blocks = mats[i].nt * mats[i].kc;
    dg_pack_w_kernel<<<blocks, 32, 0, stream>>>(
        (const float*)d_in[mats[i].idx], Wb + mats[i].off * 512,
        mats[i].K, mats[i].N, mats[i].kc);
  }
  dg_init_state_kernel<<<(3 * SO_TOT + 255) / 256, 256, 0, stream>>>(state);

  const float* mec = (const float*)d_in[16];
  float* out = (float*)d_out;
  for (int t = 0; t < TSTEPS; ++t) {
    dg_rates_pack_kernel<<<ACH_TOT, 32, 0, stream>>>(state, mec, Ab, out, t);
    dg_step_u1_kernel<<<U1_BLOCKS, 32, 0, stream>>>(Wb, Ab, state, pbuf);
    dg_step_u2_kernel<<<U2_BLOCKS, 32, 0, stream>>>(pbuf, state);
  }
}
